// KANGraphAttentionLayerARC5_76124000354753
// MI455X (gfx1250) — matrix operand order repaired and checked by reading, not compiled, not run on hardware
//
#include <hip/hip_runtime.h>
#include <stdint.h>

typedef __attribute__((ext_vector_type(16))) _Float16 v16h;
typedef __attribute__((ext_vector_type(8)))  _Float16 v8h;
typedef __attribute__((ext_vector_type(8)))  float    v8f;

#define N_NODES 50000
#define E_EDGES 800000
#define IN_F    128
#define OUT_F   64
#define KSPLINE 1024
#define KTOT    1152          // 1024 spline + 128 base (silu)
#define KSTEPS  36            // 1152 / 32
#define WPAD    1160          // padded K stride of W in LDS (bank-conflict avoidance)
#define HPAD    129           // padded row stride of h tile in LDS
#define GSTEP   (4.0f / 7.0f) // linspace(-2,2,8) step
#define INV_DEN (7.0f / 4.0f) // 1/DENOM

union V16U { v16h v; v8h h[2]; };

// ---------------- kernel: convert concat weight matrix to f16 ----------------
__global__ void k_convert_w(const float* __restrict__ spline_w,
                            const float* __restrict__ base_w,
                            _Float16* __restrict__ W16) {
  int idx = blockIdx.x * 256 + threadIdx.x;
  if (idx >= OUT_F * KTOT) return;
  int o = idx / KTOT, k = idx - o * KTOT;
  float v = (k < KSPLINE) ? spline_w[o * KSPLINE + k]
                          : base_w[o * IN_F + (k - KSPLINE)];
  W16[idx] = (_Float16)v;
}

// ---------------- kernel: zero out, init emax/denom ----------------
__global__ void k_init(float* __restrict__ out, float* __restrict__ emax,
                       float* __restrict__ denom) {
  int idx = blockIdx.x * 256 + threadIdx.x;
  if (idx < N_NODES * OUT_F) out[idx] = 0.0f;
  if (idx < N_NODES) { emax[idx] = -__builtin_inff(); denom[idx] = 0.0f; }
}

// ---- A-fragment builders (16-bit A layout: lane(g,m), halves 0..7 -> K=kbase+g*8+j,
// ---- halves 8..15 -> K=kbase+16+g*8+j) ----
__device__ __forceinline__ v16h build_a_spline(const float* __restrict__ sHm,
                                               int ks, int g) {
  const float h0 = sHm[ks * 4 + g];
  const float h1 = sHm[ks * 4 + g + 2];
  V16U a;
  #pragma unroll
  for (int j = 0; j < 8; ++j) {
    const float gj = -2.0f + (float)j * GSTEP;
    const float d0 = (h0 - gj) * INV_DEN;
    const float d1 = (h1 - gj) * INV_DEN;
    a.v[j]     = (_Float16)__expf(-d0 * d0);
    a.v[j + 8] = (_Float16)__expf(-d1 * d1);
  }
  return a.v;
}

__device__ __forceinline__ v16h build_a_silu(const float* __restrict__ sHm,
                                             int t, int g) {
  V16U a;
  #pragma unroll
  for (int e = 0; e < 16; ++e) {
    const int i = t * 32 + (e & 7) + g * 8 + ((e >> 3) << 4);
    const float hv = sHm[i];
    a.v[e] = (_Float16)(hv / (1.0f + __expf(-hv)));
  }
  return a.v;
}

__device__ __forceinline__ v16h load_b(const _Float16* __restrict__ sW,
                                       int m, int nt, int kb) {
  V16U b;
  const _Float16* wp = sW + (nt * 16 + m) * WPAD + kb;
  b.h[0] = *(const v8h*)(wp);
  b.h[1] = *(const v8h*)(wp + 16);
  return b.v;
}

// ---------------- kernel: fused KAN GEMM via WMMA ----------------
// 8 waves/block; each wave owns a 16-node tile and all 64 outputs
// (4 x v_wmma_f32_16x16x32_f16 accumulators), K = 1152, software-pipelined A.
__global__ __launch_bounds__(256)
void k_kan_gemm(const float* __restrict__ h, const _Float16* __restrict__ W16,
                const float* __restrict__ av,
                float* __restrict__ HW, float* __restrict__ s1,
                float* __restrict__ s2) {
  extern __shared__ char smem[];
  _Float16* sW = (_Float16*)smem;                              // 64 x WPAD halves
  float*    sH = (float*)(smem + (size_t)OUT_F * WPAD * 2);    // 8 x 16 x HPAD floats

  const int tid  = threadIdx.x;
  const int wave = tid >> 5;
  const int lane = tid & 31;
  const int g    = lane >> 4;   // half-wave group (ISA A/B layout)
  const int m    = lane & 15;

  // Stage entire f16 W (64x1152) into padded LDS: whole block cooperates.
  {
    const uint32_t* src = (const uint32_t*)W16;
    for (int o = 0; o < OUT_F; ++o) {
      uint32_t* dst = (uint32_t*)(sW + o * WPAD);
      for (int k2 = tid; k2 < (KTOT >> 1); k2 += 256)
        dst[k2] = src[o * (KTOT >> 1) + k2];
    }
  }

  const int  tile   = blockIdx.x * 8 + wave;
  const bool active = (tile * 16) < N_NODES;   // N = 3125*16, uniform per wave
  if (active) {
    // Coalesced stage of this wave's 16x128 h tile.
    const float* hrow = h + (size_t)tile * 16 * IN_F;
    float* sHw = sH + wave * (16 * HPAD);
    for (int t = 0; t < 64; ++t) {
      int idx = t * 32 + lane;                 // 0..2047
      sHw[(idx >> 7) * HPAD + (idx & 127)] = hrow[idx];
    }
  }
  __syncthreads();
  if (!active) return;

  const float* sHm = sH + wave * (16 * HPAD) + m * HPAD;  // this lane's node row

  v8f acc[4] = {v8f{}, v8f{}, v8f{}, v8f{}};

  // Software-pipelined K loop: issue all 8 ds_load_b128 for this step, then
  // build next step's A fragment (v_exp TRANS ops co-execute with WMMA),
  // then issue the 4 back-to-back WMMAs.
  v16h a_cur = build_a_spline(sHm, 0, g);
  #pragma unroll 1
  for (int ks = 0; ks < KSTEPS; ++ks) {
    const int kb = ks * 32 + g * 8;   // B layout: half e<8 -> K=kbase+g*8+e, e>=8 -> K=kbase+16+g*8+(e-8), N=m
    const v16h b0 = load_b(sW, m, 0, kb);
    const v16h b1 = load_b(sW, m, 1, kb);
    const v16h b2 = load_b(sW, m, 2, kb);
    const v16h b3 = load_b(sW, m, 3, kb);

    v16h a_next = a_cur;
    if (ks < KSTEPS - 1)
      a_next = (ks < 31) ? build_a_spline(sHm, ks + 1, g)
                         : build_a_silu(sHm, ks - 31, g);

    acc[0] = __builtin_amdgcn_wmma_f32_16x16x32_f16(false, a_cur, false, b0,
                                                    (short)0, acc[0], false, false);
    acc[1] = __builtin_amdgcn_wmma_f32_16x16x32_f16(false, a_cur, false, b1,
                                                    (short)0, acc[1], false, false);
    acc[2] = __builtin_amdgcn_wmma_f32_16x16x32_f16(false, a_cur, false, b2,
                                                    (short)0, acc[2], false, false);
    acc[3] = __builtin_amdgcn_wmma_f32_16x16x32_f16(false, a_cur, false, b3,
                                                    (short)0, acc[3], false, false);
    a_cur = a_next;
  }

  // ---- epilogue 1: store HW. C/D layout: VGPR r -> row M = r + 8*g, N = m.
  #pragma unroll
  for (int r = 0; r < 8; ++r) {
    const int node = tile * 16 + r + g * 8;
    float* dst = HW + (size_t)node * OUT_F + m;
    dst[0]  = acc[0][r];
    dst[16] = acc[1][r];
    dst[32] = acc[2][r];
    dst[48] = acc[3][r];
  }

  // ---- epilogue 2 (fused): s1 = HW.a1, s2 = HW.a2 via half-wave reductions.
  float a1w[4], a2w[4];
  #pragma unroll
  for (int nt = 0; nt < 4; ++nt) {
    a1w[nt] = av[nt * 16 + m];
    a2w[nt] = av[OUT_F + nt * 16 + m];
  }
  #pragma unroll
  for (int r = 0; r < 8; ++r) {
    float p1 = acc[0][r] * a1w[0] + acc[1][r] * a1w[1] +
               acc[2][r] * a1w[2] + acc[3][r] * a1w[3];
    float p2 = acc[0][r] * a2w[0] + acc[1][r] * a2w[1] +
               acc[2][r] * a2w[2] + acc[3][r] * a2w[3];
    #pragma unroll
    for (int off = 8; off >= 1; off >>= 1) {
      p1 += __shfl_xor(p1, off, 16);
      p2 += __shfl_xor(p2, off, 16);
    }
    if (m == 0) {
      const int node = tile * 16 + r + g * 8;
      s1[node] = p1;
      s2[node] = p2;
    }
  }
}

__device__ __forceinline__ void atomicMaxF(float* addr, float val) {
  // IEEE ordering trick; emax initialized to -inf.
  if (val >= 0.0f) atomicMax((int*)addr, __float_as_int(val));
  else             atomicMin((unsigned int*)addr, __float_as_uint(val));
}

// ---------------- kernel: edge scores + segment max ----------------
__global__ void k_edge_scores(const int* __restrict__ ei,
                              const float* __restrict__ s1,
                              const float* __restrict__ s2,
                              float* __restrict__ ebuf, float* __restrict__ emax) {
  int e = blockIdx.x * 256 + threadIdx.x;
  if (e >= E_EDGES) return;
  int r = ei[e], c = ei[E_EDGES + e];
  float v = s1[r] + s2[c];
  ebuf[e] = v;
  atomicMaxF(emax + r, v);
}

// ---------------- kernel: exp + segment sum ----------------
__global__ void k_edge_exp(const int* __restrict__ ei, float* __restrict__ ebuf,
                           const float* __restrict__ emax, float* __restrict__ denom) {
  int e = blockIdx.x * 256 + threadIdx.x;
  if (e >= E_EDGES) return;
  int r = ei[e];
  float v = __expf(ebuf[e] - emax[r]);
  ebuf[e] = v;
  atomicAdd(denom + r, v);
}

// ---------------- kernel: weighted scatter (16 lanes per edge, float4) --------
__global__ void k_edge_scatter(const int* __restrict__ ei,
                               const float* __restrict__ ebuf,
                               const float* __restrict__ denom,
                               const float* __restrict__ HW,
                               float* __restrict__ out) {
  int gid = blockIdx.x * 256 + threadIdx.x;   // E*16 = 12.8M
  int e = gid >> 4, q = (gid & 15) * 4;
  int r = ei[e], c = ei[E_EDGES + e];
  float attn = ebuf[e] / denom[r];
  const float4 hv = *(const float4*)(HW + (size_t)c * OUT_F + q);
  float* dst = out + (size_t)r * OUT_F + q;
  atomicAdd(dst + 0, attn * hv.x);
  atomicAdd(dst + 1, attn * hv.y);
  atomicAdd(dst + 2, attn * hv.z);
  atomicAdd(dst + 3, attn * hv.w);
}

extern "C" void kernel_launch(void* const* d_in, const int* in_sizes, int n_in,
                              void* d_out, int out_size, void* d_ws, size_t ws_size,
                              hipStream_t stream) {
  const float* h        = (const float*)d_in[0];
  const int*   ei       = (const int*)d_in[1];
  const float* spline_w = (const float*)d_in[2];
  const float* base_w   = (const float*)d_in[3];
  const float* av       = (const float*)d_in[4];
  float* out = (float*)d_out;

  // workspace layout (bytes)
  char* ws = (char*)d_ws;
  float*    HW    = (float*)(ws + 0);           // N*64*4     = 12,800,000
  float*    s1    = (float*)(ws + 12800000);    // N*4
  float*    s2    = (float*)(ws + 13000000);    // N*4
  float*    ebuf  = (float*)(ws + 13200000);    // E*4        = 3,200,000
  float*    emax  = (float*)(ws + 16400000);    // N*4
  float*    denom = (float*)(ws + 16600000);    // N*4
  _Float16* W16   = (_Float16*)(ws + 16800000); // 64*1152*2  = 147,456

  k_convert_w<<<(OUT_F * KTOT + 255) / 256, 256, 0, stream>>>(spline_w, base_w, W16);
  k_init<<<(N_NODES * OUT_F + 255) / 256, 256, 0, stream>>>(out, emax, denom);

  const size_t lds = (size_t)OUT_F * WPAD * sizeof(_Float16)   // 148,480
                   + (size_t)8 * 16 * HPAD * sizeof(float);    //  66,048  -> 214,528 < 320K
  k_kan_gemm<<<(N_NODES / 16 + 7) / 8, 256, lds, stream>>>(h, W16, av, HW, s1, s2);

  k_edge_scores<<<(E_EDGES + 255) / 256, 256, 0, stream>>>(ei, s1, s2, ebuf, emax);
  k_edge_exp<<<(E_EDGES + 255) / 256, 256, 0, stream>>>(ei, ebuf, emax, denom);
  k_edge_scatter<<<(E_EDGES * 16) / 256, 256, 0, stream>>>(ei, ebuf, denom, HW, out);
}
